// SkipGram_41644002902646
// MI455X (gfx1250) — matrix operand order repaired and checked by reading, not compiled, not run on hardware
//
#include <hip/hip_runtime.h>

typedef __attribute__((ext_vector_type(16))) __bf16 v16bf;
typedef __attribute__((ext_vector_type(8)))  __bf16 v8bf;
typedef __attribute__((ext_vector_type(8)))  float  v8f;

#define VOCAB  50000
#define EMBED  128
#define NOUT   200000   // CONTEXT * VOCAB
#define BATCH  1024
#define NCOLS  64       // output columns per GEMM block
#define LPITCH 136      // LDS pitch in bf16 (128 + 8 pad -> conflict-free ds_load_b128)

// ---------- fragment loaders ----------
// A-matrix 16x32 bf16 layout (ISA 7.12.2):
//   lanes 0-15 : elems 0-7 -> K=0..7,  elems 8-15 -> K=16..23
//   lanes 16-31: elems 0-7 -> K=8..15, elems 8-15 -> K=24..31
// caller passes pointer at (row, ks*32 + half*8); we read 8 elems, skip 8, read 8.
__device__ __forceinline__ v16bf load_a_frag(const __bf16* p) {
  const v8bf* q = reinterpret_cast<const v8bf*>(p);
  v8bf lo = q[0];
  v8bf hi = q[2];     // +16 elements
  v16bf r;
#pragma unroll
  for (int i = 0; i < 8; ++i) { r[i] = lo[i]; r[i + 8] = hi[i]; }
  return r;
}

// B-matrix 32x16 bf16 layout (matching A): lane holds column n = lane&15,
// lanes 0-15 -> K=0..7,16..23; lanes 16-31 -> K=8..15,24..31.
// caller passes pointer at (col, ks*32 + half*8); we read 8 elems, skip 8, read 8.
__device__ __forceinline__ v16bf load_b_frag(const __bf16* p) {
  const v8bf* q = reinterpret_cast<const v8bf*>(p);
  v8bf lo = q[0];
  v8bf hi = q[2];
  v16bf r;
#pragma unroll
  for (int i = 0; i < 8; ++i) { r[i] = lo[i]; r[i + 8] = hi[i]; }
  return r;
}

// ---------- kernel 1: gather column idx[b] of W1, relu, cvt bf16 ----------
__global__ void gather_relu_bf16(const int* __restrict__ idx,
                                 const float* __restrict__ W1,
                                 __bf16* __restrict__ Hbf,
                                 float* __restrict__ rowsum) {
  const int b = blockIdx.x;     // batch element
  const int e = threadIdx.x;    // embed index (128 threads)
  const int v = idx[b];
  float x = W1[(size_t)e * VOCAB + v];   // W1 is [EMBED, VOCAB] row-major
  x = fmaxf(x, 0.0f);
  Hbf[b * EMBED + e] = (__bf16)x;
  if (e == 0) rowsum[b] = 0.0f;          // zero exp-sum accumulator every call
}

// ---------- kernel 2: bf16 WMMA GEMM + logits write + exp-sum ----------
__global__ void __launch_bounds__(256)
gemm_wmma(const __bf16* __restrict__ Hbf,
          const float* __restrict__ W2,
          float* __restrict__ out,
          float* __restrict__ rowsum) {
  __shared__ __bf16 Bsm[NCOLS * LPITCH];   // 17408 bytes

  const int n0 = blockIdx.x * NCOLS;       // first output column of this block
  const int t  = threadIdx.x;

  // Stage this block's W2 chunk into LDS as bf16 (coalesced fp32 reads).
  for (int i = t; i < NCOLS * EMBED; i += 256) {
    const int col = i >> 7;       // 0..63
    const int k   = i & 127;      // 0..127
    const float w = W2[(size_t)(n0 + col) * EMBED + k];
    Bsm[col * LPITCH + k] = (__bf16)w;
  }
  __syncthreads();

  const int wave = t >> 5;
  const int lane = t & 31;
  const int half = lane >> 4;    // 0: lanes 0-15, 1: lanes 16-31
  const int ln   = lane & 15;

#pragma unroll 1
  for (int mt = 0; mt < 8; ++mt) {
    const int mtile = wave * 8 + mt;          // 0..63 (covers all 1024 batch rows)
    const int arow  = mtile * 16 + ln;        // A row held by this lane

    v8f acc[4];
#pragma unroll
    for (int nt = 0; nt < 4; ++nt)
#pragma unroll
      for (int r = 0; r < 8; ++r) acc[nt][r] = 0.0f;

#pragma unroll
    for (int ks = 0; ks < 4; ++ks) {          // K = 128 in 4 steps of 32
      const int ka = ks * 32 + half * 8;      // A-fragment base K
      const int kb = ks * 32 + half * 8;      // B-fragment base K
      const v16bf a = load_a_frag(Hbf + (size_t)arow * EMBED + ka);
#pragma unroll
      for (int nt = 0; nt < 4; ++nt) {
        const v16bf b = load_b_frag(&Bsm[(nt * 16 + ln) * LPITCH + kb]);
        acc[nt] = __builtin_amdgcn_wmma_f32_16x16x32_bf16(
            false, a, false, b, (short)0, acc[nt], false, false);
      }
    }

    // Epilogue: lane holds D rows m = r + half*8, column n = ln (per tile).
    float partial[8];
#pragma unroll
    for (int r = 0; r < 8; ++r) partial[r] = 0.0f;

#pragma unroll
    for (int nt = 0; nt < 4; ++nt) {
      const size_t cbase = (size_t)(n0 + nt * 16 + ln);
#pragma unroll
      for (int r = 0; r < 8; ++r) {
        const float lgt = acc[nt][r];
        const int row = mtile * 16 + half * 8 + r;
        out[(size_t)row * NOUT + cbase] = lgt;
        partial[r] += __expf(lgt);
      }
    }

    // Reduce exp-partials across the 16 column lanes of each half-wave.
#pragma unroll
    for (int r = 0; r < 8; ++r) {
      float p = partial[r];
      p += __shfl_xor(p, 1, 32);
      p += __shfl_xor(p, 2, 32);
      p += __shfl_xor(p, 4, 32);
      p += __shfl_xor(p, 8, 32);
      if (ln == 0) {
        const int row = mtile * 16 + half * 8 + r;
        atomicAdd(&rowsum[row], p);
      }
    }
  }
}

// ---------- kernel 3: rowsum -> log(rowsum) ----------
__global__ void log_rowsum(float* __restrict__ rs) {
  const int b = blockIdx.x * blockDim.x + threadIdx.x;
  if (b < BATCH) rs[b] = __logf(rs[b]);
}

// ---------- kernel 4: out = lse[row] - out, float4-vectorized in place ----------
__global__ void finalize(float4* __restrict__ out4, const float* __restrict__ rs) {
  const unsigned i = blockIdx.x * blockDim.x + threadIdx.x;
  const unsigned total4 = (unsigned)BATCH * (NOUT / 4);   // 51,200,000
  if (i >= total4) return;
  const unsigned row = i / (NOUT / 4);
  const float l = rs[row];
  float4 v = out4[i];
  v.x = l - v.x; v.y = l - v.y; v.z = l - v.z; v.w = l - v.w;
  out4[i] = v;
}

extern "C" void kernel_launch(void* const* d_in, const int* in_sizes, int n_in,
                              void* d_out, int out_size, void* d_ws, size_t ws_size,
                              hipStream_t stream) {
  (void)in_sizes; (void)n_in; (void)out_size; (void)ws_size;
  const int*   idx = (const int*)d_in[0];
  const float* W1  = (const float*)d_in[1];
  const float* W2  = (const float*)d_in[2];
  float* out = (float*)d_out;

  __bf16* Hbf    = (__bf16*)d_ws;                                       // 256 KB
  float*  rowsum = (float*)((char*)d_ws + (size_t)BATCH * EMBED * 2);   // 4 KB

  gather_relu_bf16<<<BATCH, EMBED, 0, stream>>>(idx, W1, Hbf, rowsum);
  gemm_wmma<<<NOUT / NCOLS, 256, 0, stream>>>(Hbf, W2, out, rowsum);
  log_rowsum<<<(BATCH + 255) / 256, 256, 0, stream>>>(rowsum);
  const unsigned total4 = (unsigned)BATCH * (NOUT / 4);
  finalize<<<(total4 + 255) / 256, 256, 0, stream>>>((float4*)out, rowsum);
}
